// Encoder_24816321036413
// MI455X (gfx1250) — matrix operand order repaired and checked by reading, not compiled, not run on hardware
//
#include <hip/hip_runtime.h>

typedef _Float16 f16;
typedef __attribute__((ext_vector_type(16))) _Float16 v16h;
typedef __attribute__((ext_vector_type(8)))  float    v8f;
typedef __attribute__((ext_vector_type(4)))  unsigned int u32x4;

union Frag16 { v16h h; u32x4 q[2]; };

__device__ __forceinline__ v8f v8f_zero() {
  v8f z = {0.f, 0.f, 0.f, 0.f, 0.f, 0.f, 0.f, 0.f};
  return z;
}

#define WMMA_F16(A_, B_, C_) \
  __builtin_amdgcn_wmma_f32_16x16x32_f16(false, (A_), false, (B_), (short)0, (C_), false, false)

// ---------------------------------------------------------------------------
// fp32 -> f16 conversion
__global__ __launch_bounds__(256) void k_cvt(const float* __restrict__ src,
                                             f16* __restrict__ dst, int n) {
  int i = blockIdx.x * 256 + threadIdx.x;
  if (i < n) dst[i] = (f16)src[i];
}

// Wo: [3][M=8][DH=512][DK=64] -> per layer [DH=512][M*DK=512]
__global__ __launch_bounds__(256) void k_cvt_wo(const float* __restrict__ src,
                                                f16* __restrict__ dst) {
  int i = blockIdx.x * 256 + threadIdx.x;          // 0 .. 3*262144
  if (i >= 3 * 262144) return;
  int n = i >> 18;
  int j = i & 262143;                              // dst index within layer
  int d = j >> 9;
  int c = j & 511;
  int m = c >> 6;
  int v = c & 63;
  dst[(size_t)n * 262144 + (size_t)d * 512 + c] =
      (f16)src[(size_t)n * 262144 + (size_t)m * 32768 + (size_t)d * 64 + v];
}

// ---------------------------------------------------------------------------
// h = x @ Wh^T + bh  (K = 2, pure VALU), dual f32/f16 output
__global__ __launch_bounds__(256) void k_inproj(const float* __restrict__ x,
                                                const float* __restrict__ Wh,
                                                const float* __restrict__ bh,
                                                float* __restrict__ h32,
                                                f16* __restrict__ h16) {
  int idx = blockIdx.x * 256 + threadIdx.x;        // over 16384*512
  if (idx >= 16384 * 512) return;
  int e = idx & 511;
  int bl = idx >> 9;
  float v = x[bl * 2 + 0] * Wh[e * 2 + 0] + x[bl * 2 + 1] * Wh[e * 2 + 1] + bh[e];
  h32[idx] = v;
  h16[idx] = (f16)v;
}

// ---------------------------------------------------------------------------
// C[M,N] = A[M,K] * Bt[N,K]^T  (+bias, +relu), A/Bt f16, C f32 and/or f16.
// Block tile 128x64, 8 waves (4 along M x 2 along N), each wave 32x32, BK=32.
__global__ __launch_bounds__(256) void k_gemm(const f16* __restrict__ A,
                                              const f16* __restrict__ Bt,
                                              float* __restrict__ Cf,
                                              f16* __restrict__ Ch,
                                              const float* __restrict__ bias,
                                              int M, int N, int K, int relu) {
  __shared__ alignas(16) f16 sA[128][40];          // 32 used + 8 pad (80B rows)
  __shared__ alignas(16) f16 sB[64][40];

  const int tid = threadIdx.x;
  const int wave = tid >> 5, lane = tid & 31;
  const int wm = wave & 3, wn = wave >> 2;
  const int lane16 = lane & 15, laneHi = lane >> 4;
  const int bm = blockIdx.y * 128;
  const int bn = blockIdx.x * 64;

  v8f acc[2][2];
#pragma unroll
  for (int i = 0; i < 2; ++i)
#pragma unroll
    for (int j = 0; j < 2; ++j) acc[i][j] = v8f_zero();

  for (int kk = 0; kk < K; kk += 32) {
    {  // stage A tile 128x32 (each thread 32B)
      int row = tid >> 1;
      int col = (tid & 1) * 16;
      const f16* src = A + (size_t)(bm + row) * K + kk + col;
      *(u32x4*)&sA[row][col]     = *(const u32x4*)src;
      *(u32x4*)&sA[row][col + 8] = *(const u32x4*)(src + 8);
    }
    {  // stage B tile 64x32 (each thread 16B)
      int row = tid >> 2;
      int col = (tid & 3) * 8;
      const f16* src = Bt + (size_t)(bn + row) * K + kk + col;
      *(u32x4*)&sB[row][col] = *(const u32x4*)src;
    }
    __syncthreads();

    Frag16 a[2], b[2];
#pragma unroll
    for (int i = 0; i < 2; ++i) {  // A frag: 16x32, rows=lanes, k per ISA layout
      const f16* p = &sA[wm * 32 + i * 16 + lane16][laneHi ? 8 : 0];
      a[i].q[0] = *(const u32x4*)p;
      a[i].q[1] = *(const u32x4*)(p + 16);
    }
#pragma unroll
    for (int j = 0; j < 2; ++j) {  // B frag: 32x16, n=lanes, k per ISA layout
      const f16* p = &sB[wn * 32 + j * 16 + lane16][laneHi ? 8 : 0];
      b[j].q[0] = *(const u32x4*)p;
      b[j].q[1] = *(const u32x4*)(p + 16);
    }
#pragma unroll
    for (int i = 0; i < 2; ++i)
#pragma unroll
      for (int j = 0; j < 2; ++j)
        acc[i][j] = WMMA_F16(a[i].h, b[j].h, acc[i][j]);
    __syncthreads();
  }

#pragma unroll
  for (int i = 0; i < 2; ++i)
#pragma unroll
    for (int j = 0; j < 2; ++j) {
      const int col = bn + wn * 32 + j * 16 + lane16;
      const float bia = bias ? bias[col] : 0.f;
#pragma unroll
      for (int r = 0; r < 8; ++r) {
        const int row = bm + wm * 32 + i * 16 + r + (laneHi ? 8 : 0);
        float v = acc[i][j][r] + bia;
        if (relu) v = v > 0.f ? v : 0.f;
        if (Cf) Cf[(size_t)row * N + col] = v;
        if (Ch) Ch[(size_t)row * N + col] = (f16)v;
      }
    }
}

// ---------------------------------------------------------------------------
// Flash attention per (b, head): Q,K,V,T laid out [B][L][M*64], head offset m*64.
// Workgroup = 4 waves x 16 q-rows = 64 rows; stream 32-wide k/v tiles.
__global__ __launch_bounds__(128) void k_attn(const f16* __restrict__ Q,
                                              const f16* __restrict__ Kk,
                                              const f16* __restrict__ V,
                                              f16* __restrict__ T) {
  __shared__ alignas(16) f16 sK[32][72];           // 32 j-rows x 64 k (+8 pad)
  __shared__ alignas(16) f16 sVt[64][40];          // transposed: [v][j]
  __shared__ alignas(16) f16 sP[4][16][40];        // per-wave P 16x32

  const int tid = threadIdx.x;
  const int wave = tid >> 5, lane = tid & 31;
  const int lane16 = lane & 15, laneHi = lane >> 4;
  const int b = blockIdx.y >> 3, m = blockIdx.y & 7;
  const int qrow0 = blockIdx.x * 64 + wave * 16;
  const size_t base = ((size_t)b * 1024) * 512 + (size_t)m * 64;
  const float scale = 0.125f;  // 1/sqrt(64)

  Frag16 aq[2];  // q strip 16x64 as two 16x32 A fragments
  {
    const f16* qp = Q + base + (size_t)(qrow0 + lane16) * 512;
#pragma unroll
    for (int kh = 0; kh < 2; ++kh) {
      const int off = kh * 32 + (laneHi ? 8 : 0);
      aq[kh].q[0] = *(const u32x4*)(qp + off);
      aq[kh].q[1] = *(const u32x4*)(qp + off + 16);
    }
  }

  float mst[8], lst[8];
  v8f accO[4];
#pragma unroll
  for (int r = 0; r < 8; ++r) { mst[r] = -1e30f; lst[r] = 0.f; }
#pragma unroll
  for (int f = 0; f < 4; ++f) accO[f] = v8f_zero();

  for (int j0 = 0; j0 < 1024; j0 += 32) {
    {  // cooperative stage: K row-major, V transposed
      const int row = tid >> 2;
      const int col = (tid & 3) * 16;
      const f16* kp = Kk + base + (size_t)(j0 + row) * 512 + col;
      const f16* vp = V + base + (size_t)(j0 + row) * 512 + col;
      *(u32x4*)&sK[row][col]     = *(const u32x4*)kp;
      *(u32x4*)&sK[row][col + 8] = *(const u32x4*)(kp + 8);
      Frag16 vv;
      vv.q[0] = *(const u32x4*)vp;
      vv.q[1] = *(const u32x4*)(vp + 8);
#pragma unroll
      for (int e = 0; e < 16; ++e) sVt[col + e][row] = vv.h[e];
    }
    __syncthreads();

    // S[16x32] = q . K_tile^T
    v8f accS[2];
    accS[0] = v8f_zero();
    accS[1] = v8f_zero();
#pragma unroll
    for (int jh = 0; jh < 2; ++jh)
#pragma unroll
      for (int kh = 0; kh < 2; ++kh) {
        Frag16 bk;  // B frag: (k, n=j): row n = sK row, k per ISA layout
        const f16* p = &sK[jh * 16 + lane16][kh * 32 + (laneHi ? 8 : 0)];
        bk.q[0] = *(const u32x4*)p;
        bk.q[1] = *(const u32x4*)(p + 16);
        accS[jh] = WMMA_F16(aq[kh].h, bk.h, accS[jh]);
      }

    // online softmax (per-row reductions across 16-lane halves)
    float corr[8];
#pragma unroll
    for (int r = 0; r < 8; ++r) {
      const float s0 = accS[0][r] * scale;
      const float s1 = accS[1][r] * scale;
      float mx = fmaxf(s0, s1);
#pragma unroll
      for (int d = 1; d < 16; d <<= 1) mx = fmaxf(mx, __shfl_xor(mx, d, 32));
      const float mnew = fmaxf(mst[r], mx);
      const float c = __expf(mst[r] - mnew);
      const float p0 = __expf(s0 - mnew);
      const float p1 = __expf(s1 - mnew);
      float ts = p0 + p1;
#pragma unroll
      for (int d = 1; d < 16; d <<= 1) ts += __shfl_xor(ts, d, 32);
      lst[r] = lst[r] * c + ts;
      mst[r] = mnew;
      corr[r] = c;
      // C layout -> row-major P in LDS (row = r + 8*laneHi, col = jh*16+lane16)
      sP[wave][r + laneHi * 8][lane16]      = (f16)p0;
      sP[wave][r + laneHi * 8][16 + lane16] = (f16)p1;
    }
#pragma unroll
    for (int f = 0; f < 4; ++f)
#pragma unroll
      for (int r = 0; r < 8; ++r) accO[f][r] *= corr[r];
    __syncthreads();  // make P (and Vt) visible for fragment gathers

    // accO += P(16x32) . V(32x64)
    Frag16 ap;
    {
      const f16* pp = &sP[wave][lane16][laneHi ? 8 : 0];
      ap.q[0] = *(const u32x4*)pp;
      ap.q[1] = *(const u32x4*)(pp + 16);
    }
#pragma unroll
    for (int vh = 0; vh < 4; ++vh) {
      Frag16 bv;  // B frag from transposed V: n = v col, k = j per ISA layout
      const f16* pv = &sVt[vh * 16 + lane16][laneHi ? 8 : 0];
      bv.q[0] = *(const u32x4*)pv;
      bv.q[1] = *(const u32x4*)(pv + 16);
      accO[vh] = WMMA_F16(ap.h, bv.h, accO[vh]);
    }
    __syncthreads();  // before next tile overwrites sK/sVt/sP
  }

  float inv[8];
#pragma unroll
  for (int r = 0; r < 8; ++r) inv[r] = 1.f / lst[r];
#pragma unroll
  for (int vh = 0; vh < 4; ++vh)
#pragma unroll
    for (int r = 0; r < 8; ++r) {
      const int row = qrow0 + r + laneHi * 8;
      const int col = vh * 16 + lane16;
      T[base + (size_t)row * 512 + col] = (f16)(accO[vh][r] * inv[r]);
    }
}

// ---------------------------------------------------------------------------
// z = A + B; per-position (l) batchnorm over (b, d) = 8192 samples;
// out = w[d]*zhat + bias[d], dual f32/f16 output. One workgroup per l.
__global__ __launch_bounds__(256) void k_bn(const float* __restrict__ A,
                                            const float* __restrict__ Bp,
                                            const float* __restrict__ wg,
                                            const float* __restrict__ bg,
                                            float* __restrict__ Out32,
                                            f16* __restrict__ Out16) {
  __shared__ float sz[8192];
  __shared__ float red[2][8];
  const int l = blockIdx.x;
  const int tid = threadIdx.x;
  float s = 0.f, s2 = 0.f;
  for (int i = tid; i < 8192; i += 256) {
    const int b = i >> 9, d = i & 511;
    const size_t idx = ((size_t)b * 1024 + l) * 512 + d;
    const float z = A[idx] + Bp[idx];
    sz[i] = z;
    s += z;
    s2 += z * z;
  }
#pragma unroll
  for (int d = 1; d < 32; d <<= 1) {
    s  += __shfl_xor(s,  d, 32);
    s2 += __shfl_xor(s2, d, 32);
  }
  const int wave = tid >> 5, lane = tid & 31;
  if (lane == 0) { red[0][wave] = s; red[1][wave] = s2; }
  __syncthreads();
  float ts = 0.f, t2 = 0.f;
#pragma unroll
  for (int wv = 0; wv < 8; ++wv) { ts += red[0][wv]; t2 += red[1][wv]; }
  const float mu = ts * (1.f / 8192.f);
  const float var = t2 * (1.f / 8192.f) - mu * mu;
  const float rstd = rsqrtf(var + 1e-5f);
  for (int i = tid; i < 8192; i += 256) {
    const int b = i >> 9, d = i & 511;
    const size_t idx = ((size_t)b * 1024 + l) * 512 + d;
    const float zn = wg[d] * ((sz[i] - mu) * rstd) + bg[d];
    Out32[idx] = zn;
    Out16[idx] = (f16)zn;
  }
}

// ---------------------------------------------------------------------------
extern "C" void kernel_launch(void* const* d_in, const int* in_sizes, int n_in,
                              void* d_out, int out_size, void* d_ws, size_t ws_size,
                              hipStream_t stream) {
  (void)in_sizes; (void)n_in; (void)out_size; (void)ws_size;
  const float* x    = (const float*)d_in[0];
  const float* Wh   = (const float*)d_in[1];
  const float* bh   = (const float*)d_in[2];
  const float* Wq   = (const float*)d_in[3];
  const float* Wk   = (const float*)d_in[4];
  const float* Wv   = (const float*)d_in[5];
  const float* Wo   = (const float*)d_in[6];
  const float* w_bn = (const float*)d_in[7];
  const float* b_bn = (const float*)d_in[8];
  const float* Wff0 = (const float*)d_in[9];
  const float* bff0 = (const float*)d_in[10];
  const float* Wff1 = (const float*)d_in[11];
  const float* bff1 = (const float*)d_in[12];
  float* out = (float*)d_out;

  char* pws = (char*)d_ws;
  auto take = [&](size_t bytes) -> void* {
    void* r = (void*)pws;
    pws += (bytes + 255) & ~(size_t)255;
    return r;
  };
  const size_t PROJ = 512 * 512;       // per-layer Wq/Wk/Wv/Wo elements
  const size_t FF   = 2048 * 512;      // per-layer Wff0/Wff1 elements
  const size_t ACT  = (size_t)16384 * 512;

  f16* Wq16 = (f16*)take(sizeof(f16) * 3 * PROJ);
  f16* Wk16 = (f16*)take(sizeof(f16) * 3 * PROJ);
  f16* Wv16 = (f16*)take(sizeof(f16) * 3 * PROJ);
  f16* Wo16 = (f16*)take(sizeof(f16) * 3 * PROJ);
  f16* Wf0h = (f16*)take(sizeof(f16) * 3 * FF);
  f16* Wf1h = (f16*)take(sizeof(f16) * 3 * FF);
  float* h32   = (float*)take(sizeof(float) * ACT);
  f16*   h16   = (f16*)take(sizeof(f16) * ACT);
  f16*   q16   = (f16*)take(sizeof(f16) * ACT);
  f16*   k16   = (f16*)take(sizeof(f16) * ACT);
  f16*   v16   = (f16*)take(sizeof(f16) * ACT);
  f16*   t16   = (f16*)take(sizeof(f16) * ACT);
  float* tmp32 = (float*)take(sizeof(float) * ACT);
  float* hbn32 = (float*)take(sizeof(float) * ACT);
  f16*   hbn16 = (f16*)take(sizeof(f16) * ACT);
  f16*   u16   = (f16*)take(sizeof(f16) * (size_t)16384 * 2048);

  const dim3 blk256(256), blk128(128);

  k_cvt<<<dim3((unsigned)((3 * PROJ + 255) / 256)), blk256, 0, stream>>>(Wq, Wq16, (int)(3 * PROJ));
  k_cvt<<<dim3((unsigned)((3 * PROJ + 255) / 256)), blk256, 0, stream>>>(Wk, Wk16, (int)(3 * PROJ));
  k_cvt<<<dim3((unsigned)((3 * PROJ + 255) / 256)), blk256, 0, stream>>>(Wv, Wv16, (int)(3 * PROJ));
  k_cvt_wo<<<dim3((unsigned)((3 * PROJ + 255) / 256)), blk256, 0, stream>>>(Wo, Wo16);
  k_cvt<<<dim3((unsigned)((3 * FF + 255) / 256)), blk256, 0, stream>>>(Wff0, Wf0h, (int)(3 * FF));
  k_cvt<<<dim3((unsigned)((3 * FF + 255) / 256)), blk256, 0, stream>>>(Wff1, Wf1h, (int)(3 * FF));

  k_inproj<<<dim3((unsigned)(ACT / 256)), blk256, 0, stream>>>(x, Wh, bh, h32, h16);

  for (int n = 0; n < 3; ++n) {
    const dim3 g512(512 / 64, 16384 / 128);
    const dim3 g2048(2048 / 64, 16384 / 128);
    // QKV projections -> f16 [B,L,M*64]
    k_gemm<<<g512, blk256, 0, stream>>>(h16, Wq16 + n * PROJ, nullptr, q16, nullptr, 16384, 512, 512, 0);
    k_gemm<<<g512, blk256, 0, stream>>>(h16, Wk16 + n * PROJ, nullptr, k16, nullptr, 16384, 512, 512, 0);
    k_gemm<<<g512, blk256, 0, stream>>>(h16, Wv16 + n * PROJ, nullptr, v16, nullptr, 16384, 512, 512, 0);
    // flash attention: grid = (L/64 row blocks, B*M heads)
    k_attn<<<dim3(1024 / 64, 16 * 8), blk128, 0, stream>>>(q16, k16, v16, t16);
    // output projection (sums heads) -> f32
    k_gemm<<<g512, blk256, 0, stream>>>(t16, Wo16 + n * PROJ, tmp32, nullptr, nullptr, 16384, 512, 512, 0);
    // BN1(h + mha)
    k_bn<<<dim3(1024), blk256, 0, stream>>>(h32, tmp32, w_bn + n * 512, b_bn + n * 512, hbn32, hbn16);
    // FFN
    k_gemm<<<g2048, blk256, 0, stream>>>(hbn16, Wf0h + n * FF, nullptr, u16, bff0 + n * 2048, 16384, 2048, 512, 1);
    k_gemm<<<g512, blk256, 0, stream>>>(u16, Wf1h + n * FF, tmp32, nullptr, bff1 + n * 512, 16384, 512, 2048, 0);
    // BN2(hbn + ff) -> next h (or final output)
    float* o32 = (n == 2) ? out : h32;
    k_bn<<<dim3(1024), blk256, 0, stream>>>(hbn32, tmp32, w_bn + n * 512, b_bn + n * 512, o32, h16);
  }
}
